// JITGNN_3461743640997
// MI455X (gfx1250) — matrix operand order repaired and checked by reading, not compiled, not run on hardware
//
#include <hip/hip_runtime.h>
#include <hip/hip_bf16.h>
#include <math.h>

typedef __bf16 bf16_t;
typedef __attribute__((ext_vector_type(16))) __bf16 v16bf;
typedef __attribute__((ext_vector_type(8)))  __bf16 v8bf;
typedef __attribute__((ext_vector_type(8)))  float  v8f;

// -------- helpers: async global->LDS copy (CDNA5), 16 bytes per lane --------
__device__ __forceinline__ unsigned lds_off(const void* p) {
  // generic pointer to LDS: low 32 bits are the wave-relative LDS byte offset
  return (unsigned)(unsigned long long)p;
}
__device__ __forceinline__ void async_copy_b128(void* lds_dst, const void* gsrc) {
  asm volatile("global_load_async_to_lds_b128 %0, %1, off"
               :: "v"(lds_off(lds_dst)), "v"((unsigned long long)gsrc)
               : "memory");
}
__device__ __forceinline__ void wait_async0() {
  asm volatile("s_wait_asynccnt 0x0" ::: "memory");
}

// ---------------------------------------------------------------- conversion
__global__ __launch_bounds__(256) void k_f32_to_bf16(const float* __restrict__ src,
                                                     bf16_t* __restrict__ dst, int n) {
  int i = blockIdx.x * 256 + threadIdx.x;
  if (i < n) dst[i] = (bf16_t)src[i];
}

// weight convert + transpose: W[K,256] f32 -> WT[256,K] bf16 ([n][k])
__global__ __launch_bounds__(256) void k_w_to_bf16_t(const float* __restrict__ W,
                                                     bf16_t* __restrict__ WT, int K) {
  int n = blockIdx.x;  // 0..255
  for (int k = threadIdx.x; k < K; k += 256)
    WT[(size_t)n * K + k] = (bf16_t)W[(size_t)k * 256 + n];
}

// ---------------------------------------------------------------- WMMA GEMM
// C[N,256] f32 = A[N,K] bf16 (row-major) @ B (given as WT[256,K], i.e. B^T)
// Block: 256 threads = 8 waves; block tile 64 rows x 256 cols.
// Wave (wr,wc): rows 16*wr..+15, cols 128*wc..+127 -> 8 accum tiles of 16x16.
__global__ __launch_bounds__(256) void k_gemm_bf16(const bf16_t* __restrict__ A,
                                                   const bf16_t* __restrict__ WT,
                                                   float* __restrict__ C,
                                                   int Nrows, int K) {
  __shared__ __align__(16) bf16_t Asl[64 * 32];    //  4 KB, row-major [r][k]
  __shared__ __align__(16) bf16_t Bsl[256 * 32];   // 16 KB, [n][k] (pre-transposed)

  const int tid  = threadIdx.x;
  const int wave = tid >> 5;
  const int lane = tid & 31;
  const int half = lane >> 4;      // 0/1: which K-half this lane holds
  const int l15  = lane & 15;
  const int rowBase = blockIdx.x * 64;
  const int wr = wave >> 1;        // 0..3
  const int wc = wave & 1;         // 0..1

  v8f acc[8];
  for (int t = 0; t < 8; ++t)
    for (int e = 0; e < 8; ++e) acc[t][e] = 0.0f;

  const int ar = tid >> 2;         // A stage: row 0..63
  const int ac = (tid & 3) * 8;    // A stage: col 0/8/16/24
  const int gr = rowBase + ar;     // may run past Nrows: reads stay inside the
                                   // workspace; those accum rows are never stored
  const int bn = tid;              // B stage: output column 0..255

  for (int k0 = 0; k0 < K; k0 += 32) {
    // ---- stage A tile (64x32) and B tile ([n][k0..k0+31]) via async DMA
    async_copy_b128(Asl + ar * 32 + ac, A + (size_t)gr * K + k0 + ac);
    {
      const bf16_t* bsrc = WT + (size_t)bn * K + k0;
      async_copy_b128(Bsl + bn * 32 + 0,  bsrc + 0);
      async_copy_b128(Bsl + bn * 32 + 8,  bsrc + 8);
      async_copy_b128(Bsl + bn * 32 + 16, bsrc + 16);
      async_copy_b128(Bsl + bn * 32 + 24, bsrc + 24);
    }
    wait_async0();
    __syncthreads();

    // ---- A fragment (ISA 16-bit A 16x32 layout):
    // lanes 0-15: K 0-7 (v0-3) & 16-23 (v4-7); lanes 16-31: K 8-15 & 24-31
    union { v16bf v; v8bf h[2]; } af;
    const bf16_t* arow = Asl + (wr * 16 + l15) * 32;
    af.h[0] = *(const v8bf*)(arow + half * 8);
    af.h[1] = *(const v8bf*)(arow + 16 + half * 8);

    // ---- 8 WMMA tiles across this wave's 128-col half
    for (int t = 0; t < 8; ++t) {
      // B 32x16 layout (matches A): lanes 0-15: K 0-7 & 16-23; lanes 16-31: K 8-15 & 24-31, column l15
      union { v16bf v; v8bf h[2]; } bfv;
      const bf16_t* bcol = Bsl + (size_t)(wc * 128 + t * 16 + l15) * 32;
      bfv.h[0] = *(const v8bf*)(bcol + half * 8);
      bfv.h[1] = *(const v8bf*)(bcol + 16 + half * 8);
      acc[t] = __builtin_amdgcn_wmma_f32_16x16x32_bf16(
          false, af.v, false, bfv.v, (short)0, acc[t], false, false);
    }
    __syncthreads();  // all waves done reading before next async overwrite
  }

  // ---- store C (32-bit C/D 16x16 layout: n=l15, m = vgpr + 8*half)
  for (int t = 0; t < 8; ++t) {
    int col = wc * 128 + t * 16 + l15;
    for (int v = 0; v < 8; ++v) {
      int row = rowBase + wr * 16 + half * 8 + v;
      if (row < Nrows) C[(size_t)row * 256 + col] = acc[t][v];
    }
  }
}

// ---------------------------------------------------------------- SpMM scatter
// one wave per edge: ACC[dst,:] += w * S[src,:]
__global__ __launch_bounds__(256) void k_spmm(const float* __restrict__ S,
                                              const int* __restrict__ src,
                                              const int* __restrict__ dst,
                                              const float* __restrict__ ew,
                                              float* __restrict__ ACC, int E) {
  int e = blockIdx.x * 8 + (threadIdx.x >> 5);
  int lane = threadIdx.x & 31;
  if (e >= E) return;
  int s = src[e], d = dst[e];
  float w = ew[e];
  const float4* sp = (const float4*)(S + (size_t)s * 256);
  float* dp = ACC + (size_t)d * 256 + lane * 8;
  float4 v0 = sp[lane * 2];
  float4 v1 = sp[lane * 2 + 1];
  atomicAdd(dp + 0, v0.x * w); atomicAdd(dp + 1, v0.y * w);
  atomicAdd(dp + 2, v0.z * w); atomicAdd(dp + 3, v0.w * w);
  atomicAdd(dp + 4, v1.x * w); atomicAdd(dp + 5, v1.y * w);
  atomicAdd(dp + 6, v1.z * w); atomicAdd(dp + 7, v1.w * w);
}

// ---------------------------------------------------------------- bias + relu
// ACC = relu(ACC + b[col]) in f32 (for pooling), also written bf16 for next GEMM
__global__ __launch_bounds__(256) void k_relu_bias(float* __restrict__ ACC,
                                                   const float* __restrict__ bias,
                                                   bf16_t* __restrict__ Hb) {
  size_t i = (size_t)blockIdx.x * 256 + threadIdx.x;
  float v = ACC[i] + bias[threadIdx.x];  // 256 cols == blockDim
  v = v > 0.0f ? v : 0.0f;
  ACC[i] = v;
  Hb[i] = (bf16_t)v;
}

// ---------------------------------------------------------------- column sums
__global__ __launch_bounds__(256) void k_colsum(const float* __restrict__ H,
                                                float* __restrict__ out, int Nr) {
  int col = threadIdx.x;
  int r0 = blockIdx.x * 256;
  int r1 = r0 + 256; if (r1 > Nr) r1 = Nr;
  float a = 0.0f;
  for (int r = r0; r < r1; ++r) a += H[(size_t)r * 256 + col];
  atomicAdd(out + col, a);
}

// g = tanh(mean @ attn_W)
__global__ __launch_bounds__(256) void k_attn_g(const float* __restrict__ colsum,
                                                const float* __restrict__ W,
                                                float* __restrict__ g, float inv) {
  __shared__ float m[256];
  int j = threadIdx.x;
  m[j] = colsum[j] * inv;
  __syncthreads();
  float a = 0.0f;
  for (int i = 0; i < 256; ++i) a += m[i] * W[i * 256 + j];
  g[j] = tanhf(a);
}

// pooled = sum_r sigmoid(H[r]·g) * H[r]   (rows 0..Nr-1)
__global__ __launch_bounds__(256) void k_attn_pool(const float* __restrict__ H,
                                                   const float* __restrict__ g,
                                                   float* __restrict__ pooled,
                                                   int Nr, int rowsPerBlock) {
  __shared__ float gs[256];
  int tid = threadIdx.x, wave = tid >> 5, lane = tid & 31;
  gs[tid] = g[tid];
  __syncthreads();
  float gl[8], acc[8];
  for (int c = 0; c < 8; ++c) { gl[c] = gs[lane * 8 + c]; acc[c] = 0.0f; }
  int r0 = blockIdx.x * rowsPerBlock;
  int r1 = r0 + rowsPerBlock; if (r1 > Nr) r1 = Nr;
  for (int r = r0 + wave; r < r1; r += 8) {
    const float* row = H + (size_t)r * 256;
    float x[8], p = 0.0f;
    for (int c = 0; c < 8; ++c) { x[c] = row[lane * 8 + c]; p += x[c] * gl[c]; }
    for (int off = 16; off > 0; off >>= 1) p += __shfl_xor(p, off, 32);
    float s = 1.0f / (1.0f + __expf(-p));
    for (int c = 0; c < 8; ++c) acc[c] += s * x[c];
  }
  for (int c = 0; c < 8; ++c) atomicAdd(pooled + lane * 8 + c, acc[c]);
}

// ---------------------------------------------------------------- final tail
__global__ __launch_bounds__(256) void k_final(const float* __restrict__ bemb,
                                               const float* __restrict__ aemb,
                                               const float* __restrict__ tnW,    // [256][256][32]
                                               const float* __restrict__ tnWb,   // [32][512]
                                               const float* __restrict__ tnBias, // [32]
                                               const float* __restrict__ fcW,    // [46]
                                               const float* __restrict__ fcb,    // [1]
                                               const float* __restrict__ metrics,// [14]
                                               float* __restrict__ out) {        // [33]
  __shared__ float bs[256], as[256], part[8][32], aggs[32];
  int tid = threadIdx.x;
  bs[tid] = bemb[tid]; as[tid] = aemb[tid];
  __syncthreads();
  int t = tid & 31, jg = tid >> 5;
  float sc = 0.0f;
  for (int j = jg; j < 256; j += 8) {
    float inner = 0.0f;
    for (int i = 0; i < 256; ++i) inner += bs[i] * tnW[(size_t)i * 8192 + j * 32 + t];
    sc += inner * as[j];
  }
  part[jg][t] = sc;
  __syncthreads();
  if (tid < 32) {
    float scoring = 0.0f;
    for (int k = 0; k < 8; ++k) scoring += part[k][tid];
    float blk = 0.0f;
    for (int k = 0; k < 256; ++k)
      blk += tnWb[tid * 512 + k] * bs[k] + tnWb[tid * 512 + 256 + k] * as[k];
    float agg = scoring + blk + tnBias[tid];
    agg = agg > 0.0f ? agg : 0.0f;
    aggs[tid] = agg;
    out[1 + tid] = agg;
  }
  __syncthreads();
  if (tid == 0) {
    float o = fcb[0];
    for (int k = 0; k < 32; ++k) o += aggs[k] * fcW[k];
    for (int m = 0; m < 14; ++m) o += metrics[m] * fcW[32 + m];
    out[0] = o;
  }
}

// ---------------------------------------------------------------- launcher
extern "C" void kernel_launch(void* const* d_in, const int* in_sizes, int n_in,
                              void* d_out, int out_size, void* d_ws, size_t ws_size,
                              hipStream_t stream) {
  (void)n_in; (void)out_size; (void)ws_size;
  const int H = 512, M = 256;
  const int N = in_sizes[0] / H;
  const int E = in_sizes[2];
  const int Nr = N - 1;

  // d_in layout: 0 b_x, 1 b_adj_idx, 2 b_adj_w, 3 a_x, 4 a_adj_idx, 5 a_adj_w,
  // 6 metrics, 7..14 g1 params (W1,b1..W4,b4), 15..22 g2 params,
  // 23 attn_W, 24 tn_W, 25 tn_Wb, 26 tn_bias, 27 fc_W, 28 fc_b
  const float* metrics = (const float*)d_in[6];
  const float* attnW   = (const float*)d_in[23];

  char* ws = (char*)d_ws;
  size_t offIn  = 0;                                   // bf16 activations [N, 512]
  size_t offS   = offIn + (size_t)N * 512 * 2;         // f32 GEMM out [N, 256]
  size_t offAcc = offS + (size_t)N * 256 * 4;          // f32 SpMM accum [N, 256]
  size_t offW   = offAcc + (size_t)N * 256 * 4;        // bf16 W^T staging (256*512)
  size_t offSm  = offW + (size_t)512 * 256 * 2;        // small vectors
  bf16_t* INb = (bf16_t*)(ws + offIn);
  float*  S   = (float*)(ws + offS);
  float*  ACC = (float*)(ws + offAcc);
  bf16_t* WT  = (bf16_t*)(ws + offW);
  float*  csum = (float*)(ws + offSm);          // 256
  float*  gbuf = csum + 256;                    // 256
  float*  emb0 = gbuf + 256;                    // 256 (b_emb)
  float*  emb1 = emb0 + 256;                    // 256 (a_emb)
  float*  embs[2] = { emb0, emb1 };

  const int gemmBlocks = (N + 63) / 64;
  const int spmmBlocks = (E + 7) / 8;

  for (int br = 0; br < 2; ++br) {
    const float* x  = (const float*)d_in[br * 3 + 0];
    const int*   ix = (const int*)  d_in[br * 3 + 1];
    const float* ew = (const float*)d_in[br * 3 + 2];
    const int*   src = ix;
    const int*   dst = ix + E;
    const int pbase = 7 + br * 8;

    // input features -> bf16
    {
      int n = N * H;
      k_f32_to_bf16<<<(n + 255) / 256, 256, 0, stream>>>(x, INb, n);
    }

    for (int l = 0; l < 4; ++l) {
      const float* Wl = (const float*)d_in[pbase + 2 * l + 0];
      const float* bl = (const float*)d_in[pbase + 2 * l + 1];
      int K = (l == 0) ? H : M;

      k_w_to_bf16_t<<<256, 256, 0, stream>>>(Wl, WT, K);
      k_gemm_bf16<<<gemmBlocks, 256, 0, stream>>>(INb, WT, S, N, K);
      hipMemsetAsync(ACC, 0, (size_t)N * 256 * 4, stream);
      k_spmm<<<spmmBlocks, 256, 0, stream>>>(S, src, dst, ew, ACC, E);
      k_relu_bias<<<N, 256, 0, stream>>>(ACC, bl, INb);
    }

    // attention pooling on rows 0..Nr-1 of ACC (final f32 embeddings)
    hipMemsetAsync(csum, 0, 256 * 4, stream);
    k_colsum<<<(Nr + 255) / 256, 256, 0, stream>>>(ACC, csum, Nr);
    k_attn_g<<<1, 256, 0, stream>>>(csum, attnW, gbuf, 1.0f / (float)Nr);
    hipMemsetAsync(embs[br], 0, 256 * 4, stream);
    k_attn_pool<<<(Nr + 1023) / 1024, 256, 0, stream>>>(ACC, gbuf, embs[br], Nr, 1024);
  }

  k_final<<<1, 256, 0, stream>>>(emb0, emb1,
                                 (const float*)d_in[24], (const float*)d_in[25],
                                 (const float*)d_in[26], (const float*)d_in[27],
                                 (const float*)d_in[28], metrics,
                                 (float*)d_out);
}
